// EnhancedGNNTransformer_33612414059123
// MI455X (gfx1250) — matrix operand order repaired and checked by reading, not compiled, not run on hardware
//
#include <hip/hip_runtime.h>
#include <hip/hip_bf16.h>
#include <math.h>

// ---------------------------------------------------------------------------
// EnhancedGNNTransformer for MI455X (gfx1250, wave32, WMMA).
// All dense GEMMs ([N,K]x[K,Nc]) run through a bf16->f32 WMMA kernel using
// v_wmma_f32_16x16x32_bf16. A tiles are staged into LDS via the gfx1250 async
// load-to-LDS path when available (ASYNCcnt), B tiles via coalesced b128 loads
// with a bank-conflict-free padded LDS transpose. Graph ops (edge softmax,
// scatter aggregation, pooling) use f32 global atomics.
// ---------------------------------------------------------------------------

#define NN   50000
#define EE   200000
#define ELP  (EE + NN)   // edges + self loops
#define GG   64
#define HIDD 128
#define LL   6

#if defined(__AMDGCN__) && __has_builtin(__builtin_amdgcn_global_load_async_to_lds_b128) && __has_builtin(__builtin_amdgcn_s_wait_asynccnt)
#define USE_ASYNC_LDS 1
#else
#define USE_ASYNC_LDS 0
#endif

typedef __attribute__((ext_vector_type(16))) __bf16 v16bf;
typedef __attribute__((ext_vector_type(8)))  __bf16 v8bf;
typedef __attribute__((ext_vector_type(8)))  float  v8f;

#if USE_ASYNC_LDS
typedef int v4i_vs __attribute__((vector_size(16)));
typedef __attribute__((address_space(1))) v4i_vs gv4i;   // global int4*
typedef __attribute__((address_space(3))) v4i_vs lv4i;   // LDS int4*
#endif

// ------------------------------ device helpers -----------------------------

__device__ __forceinline__ void atomicMaxF(float* addr, float val) {
  int* ia = (int*)addr;
  int old = __float_as_int(*addr);
  while (__int_as_float(old) < val) {
    int assumed = old;
    old = atomicCAS(ia, assumed, __float_as_int(val));
    if (old == assumed) break;
  }
}

__device__ __forceinline__ float eluf(float v) {
  return v > 0.f ? v : (expf(v) - 1.f);
}

__device__ __forceinline__ void edge_sd(const int* ei, int eidx, int& s, int& d) {
  if (eidx < EE) { s = ei[eidx]; d = ei[EE + eidx]; }
  else           { s = eidx - EE; d = s; }            // self loop
}

// ------------------------------ utility kernels ----------------------------

__global__ void fill_kernel(float* __restrict__ p, float v, long n) {
  long i = (long)blockIdx.x * blockDim.x + threadIdx.x;
  if (i < n) p[i] = v;
}

__global__ void cast_bf16_kernel(const float* __restrict__ in, __bf16* __restrict__ out, long n) {
  long i = (long)blockIdx.x * blockDim.x + threadIdx.x;
  if (i < n) out[i] = (__bf16)in[i];
}

__global__ void bias_elu_kernel(float* __restrict__ x, const float* __restrict__ b,
                                long n, int cols) {
  long i = (long)blockIdx.x * blockDim.x + threadIdx.x;
  if (i >= n) return;
  int j = (int)(i % cols);
  x[i] = eluf(x[i] + b[j]);
}

// ------------------------------ WMMA GEMM ----------------------------------
// C[M,Nc] = act( A[M,K](bf16) @ B[K,Nc](bf16) + bias[Nc] ), act: 0=none 1=gelu
// Block tile 128(M) x 64(N), K step 32. 256 threads = 8 waves; wave w computes
// rows [16w,16w+16) x 64 cols via 4 accumulators of v_wmma_f32_16x16x32_bf16.
#define BT_PAD 40   // padded k-stride of transposed B tile: 80B = 20 banks
__global__ __launch_bounds__(256)
void gemm_bf16_wmma_kernel(const __bf16* __restrict__ A, const __bf16* __restrict__ B,
                           const float* __restrict__ bias, float* __restrict__ C,
                           int M, int K, int Nc, int act) {
  __shared__ __align__(64) __bf16 As[128][32];       // row-major A tile
  __shared__ __align__(64) __bf16 Bt[64][BT_PAD];    // B tile, transposed: Bt[n][k]

  const int tid  = threadIdx.x;
  const int wave = tid >> 5;
  const int lane = tid & 31;
  const int m0   = blockIdx.x * 128;
  const int n0   = blockIdx.y * 64;

  v8f acc[4] = {};

  // A staging: thread -> (row, 16-element half); rows clamped to M-1 (a clamped
  // duplicate row only feeds output rows >= M, which are never stored).
  const int aRow  = tid >> 1;
  const int aHalf = tid & 1;
  const int gr    = min(m0 + aRow, M - 1);
  const __bf16* aSrc = A + (size_t)gr * K + aHalf * 16;

  // B staging: thread -> (k row, 8 contiguous n): one b128 load, 8 b16 LDS
  // stores into distinct banks (row stride 20 banks).
  const int bK = tid >> 3;         // 0..31
  const int bN = (tid & 7) * 8;    // 0..56

  const int r  = lane & 15;
  const int hi = lane >> 4;

  for (int k0 = 0; k0 < K; k0 += 32) {
    // ---- stage A tile ----
#if USE_ASYNC_LDS
    {
      gv4i* g = (gv4i*)(aSrc + k0);
      lv4i* l = (lv4i*)&As[aRow][aHalf * 16];
      __builtin_amdgcn_global_load_async_to_lds_b128(g, l, 0, 0);
      __builtin_amdgcn_global_load_async_to_lds_b128(g, l, 16, 0);
    }
#else
    {
      const uint4* src = (const uint4*)(aSrc + k0);
      uint4* dst = (uint4*)&As[aRow][aHalf * 16];
      dst[0] = src[0];
      dst[1] = src[1];
    }
#endif
    if (k0 + 32 < K)  // hint next K tile (global_prefetch_b8)
      __builtin_prefetch(aSrc + k0 + 32, 0, 3);

    // ---- stage B tile (coalesced b128 per thread, transposed scatter) ----
    {
      union { uint4 u; __bf16 e[8]; } bvv;
      bvv.u = *(const uint4*)(B + (size_t)(k0 + bK) * Nc + n0 + bN);
      #pragma unroll
      for (int i = 0; i < 8; ++i) Bt[bN + i][bK] = bvv.e[i];
      if (k0 + 32 < K)
        __builtin_prefetch(B + (size_t)(k0 + 32 + bK) * Nc + n0 + bN, 0, 3);
    }
#if USE_ASYNC_LDS
    __builtin_amdgcn_s_wait_asynccnt(0);
#endif
    __syncthreads();

    // ---- A fragment per ISA 16-bit A layout:
    // lanes 0-15: K 0..7 & 16..23 ; lanes 16-31: K 8..15 & 24..31
    union { v16bf v; v8bf h[2]; } af;
    af.h[0] = *(const v8bf*)&As[wave * 16 + r][hi * 8];
    af.h[1] = *(const v8bf*)&As[wave * 16 + r][16 + hi * 8];

    // ---- B fragments: lanes 0-15 hold K 0..7 & 16..23, lanes 16-31 K 8..15 & 24..31, col = lane&15
    #pragma unroll
    for (int nt = 0; nt < 4; ++nt) {
      union { v16bf v; v8bf h[2]; } bf;
      bf.h[0] = *(const v8bf*)&Bt[nt * 16 + r][hi * 8];
      bf.h[1] = *(const v8bf*)&Bt[nt * 16 + r][16 + hi * 8];
      acc[nt] = __builtin_amdgcn_wmma_f32_16x16x32_bf16(
          false, af.v, false, bf.v, (short)0, acc[nt], false, false);
    }
    __syncthreads();
  }

  // ---- epilogue: C/D layout -> row = 16*wave + 8*hi + i, col = n0+16*nt+r ----
  #pragma unroll
  for (int nt = 0; nt < 4; ++nt) {
    const int col = n0 + nt * 16 + r;
    const float bv = bias ? bias[col] : 0.f;
    #pragma unroll
    for (int i = 0; i < 8; ++i) {
      const int row = m0 + wave * 16 + hi * 8 + i;
      if (row < M) {
        float v = acc[nt][i] + bv;
        if (act == 1) v = 0.5f * v * (1.0f + erff(v * 0.70710678118654752f));
        C[(size_t)row * Nc + col] = v;
      }
    }
  }
}

// ------------------------------ GAT kernels --------------------------------

// per (node, head): asrc = <h[n,hh,:], a_s[hh,:]>, adst likewise  (C=128)
__global__ void gat_alpha_kernel(const float* __restrict__ hfeat,
                                 const float* __restrict__ a_s, const float* __restrict__ a_d,
                                 float* __restrict__ asrc, float* __restrict__ adst, int H) {
  int idx = blockIdx.x * blockDim.x + threadIdx.x;
  if (idx >= NN * H) return;
  int node = idx / H, hh = idx % H;
  const float* f = hfeat + (size_t)node * H * HIDD + (size_t)hh * HIDD;
  const float* s = a_s + hh * HIDD;
  const float* d = a_d + hh * HIDD;
  float s0 = 0.f, s1 = 0.f;
  for (int c = 0; c < HIDD; ++c) { float v = f[c]; s0 += v * s[c]; s1 += v * d[c]; }
  asrc[idx] = s0;
  adst[idx] = s1;
}

__global__ void edge_score_kernel(const int* __restrict__ ei, int H,
                                  const float* __restrict__ asrc, const float* __restrict__ adst,
                                  float* __restrict__ esc, float* __restrict__ mx) {
  int idx = blockIdx.x * blockDim.x + threadIdx.x;
  if (idx >= ELP * H) return;
  int eidx = idx / H, hh = idx % H;
  int s, d; edge_sd(ei, eidx, s, d);
  float v = asrc[s * H + hh] + adst[d * H + hh];
  v = v > 0.f ? v : 0.2f * v;   // leaky_relu(0.2)
  esc[idx] = v;
  atomicMaxF(&mx[d * H + hh], v);
}

__global__ void edge_exp_kernel(const int* __restrict__ ei, int H,
                                float* __restrict__ esc, const float* __restrict__ mx,
                                float* __restrict__ sm) {
  int idx = blockIdx.x * blockDim.x + threadIdx.x;
  if (idx >= ELP * H) return;
  int eidx = idx / H, hh = idx % H;
  int s, d; edge_sd(ei, eidx, s, d);
  (void)s;
  float mm = mx[d * H + hh];
  if (!(fabsf(mm) < 1e30f)) mm = 0.f;   // isfinite guard (matches reference)
  float ex = expf(esc[idx] - mm);
  esc[idx] = ex;
  atomicAdd(&sm[d * H + hh], ex);
}

// scatter out[dst, j] += alpha(e, j/128) * feat[src, j]   (block per edge)
__global__ void edge_aggr_kernel(const int* __restrict__ ei, int H,
                                 const float* __restrict__ esc, const float* __restrict__ sm,
                                 const float* __restrict__ feat, float* __restrict__ out) {
  int eidx = blockIdx.x;
  int s, d; edge_sd(ei, eidx, s, d);
  const int HC = H * HIDD;
  for (int j = threadIdx.x; j < HC; j += blockDim.x) {
    int hh = j >> 7;                     // C = 128
    float alpha = esc[eidx * H + hh] / (sm[d * H + hh] + 1e-16f);
    atomicAdd(&out[(size_t)d * HC + j], alpha * feat[(size_t)s * HC + j]);
  }
}

// ------------------------------ GCN kernels --------------------------------

__global__ void deg_kernel(const int* __restrict__ ei, float* __restrict__ deg) {
  int e = blockIdx.x * blockDim.x + threadIdx.x;
  if (e >= ELP) return;
  int s, d; edge_sd(ei, e, s, d);
  (void)s;
  atomicAdd(&deg[d], 1.0f);
}

__global__ void dinv_kernel(const float* __restrict__ deg, float* __restrict__ dinv) {
  int i = blockIdx.x * blockDim.x + threadIdx.x;
  if (i >= NN) return;
  float dg = deg[i];
  dinv[i] = dg > 0.f ? rsqrtf(fmaxf(dg, 1e-12f)) : 0.f;
}

__global__ void gcn_aggr_kernel(const int* __restrict__ ei, const float* __restrict__ dinv,
                                const float* __restrict__ hw, float* __restrict__ out) {
  int eidx = blockIdx.x;
  int s, d; edge_sd(ei, eidx, s, d);
  float nrm = dinv[s] * dinv[d];
  for (int j = threadIdx.x; j < HIDD; j += blockDim.x)
    atomicAdd(&out[(size_t)d * HIDD + j], nrm * hw[(size_t)s * HIDD + j]);
}

// ------------------------------ LayerNorm ----------------------------------
// h = LN(h + add) over last dim 128; one wave32 per row, lane owns 4 elements.
__global__ void ln_add_kernel(float* __restrict__ h, const float* __restrict__ add,
                              const float* __restrict__ g, const float* __restrict__ b) {
  int row  = blockIdx.x * 8 + (threadIdx.x >> 5);
  int lane = threadIdx.x & 31;
  if (row >= NN) return;
  float x[4];
  float sum = 0.f;
  #pragma unroll
  for (int k = 0; k < 4; ++k) {
    int j = lane + 32 * k;
    x[k] = h[(size_t)row * HIDD + j] + add[(size_t)row * HIDD + j];
    sum += x[k];
  }
  #pragma unroll
  for (int o = 16; o > 0; o >>= 1) sum += __shfl_xor(sum, o, 32);
  float mu = sum * (1.f / 128.f);
  float vs = 0.f;
  #pragma unroll
  for (int k = 0; k < 4; ++k) { float dd = x[k] - mu; vs += dd * dd; }
  #pragma unroll
  for (int o = 16; o > 0; o >>= 1) vs += __shfl_xor(vs, o, 32);
  float rinv = rsqrtf(vs * (1.f / 128.f) + 1e-5f);
  #pragma unroll
  for (int k = 0; k < 4; ++k) {
    int j = lane + 32 * k;
    h[(size_t)row * HIDD + j] = (x[k] - mu) * rinv * g[j] + b[j];
  }
}

// ------------------------------ pooling + head -----------------------------

__global__ void pool_kernel(const float* __restrict__ h, const int* __restrict__ batch,
                            float* __restrict__ hm, float* __restrict__ hx,
                            float* __restrict__ cnt) {
  int node = blockIdx.x;
  int g = batch[node];
  for (int j = threadIdx.x; j < HIDD; j += blockDim.x) {
    float v = h[(size_t)node * HIDD + j];
    atomicAdd(&hm[g * HIDD + j], v);
    atomicMaxF(&hx[g * HIDD + j], v);
  }
  if (threadIdx.x == 0) atomicAdd(&cnt[g], 1.0f);
}

__global__ void head_kernel(const float* __restrict__ hm, const float* __restrict__ hx,
                            const float* __restrict__ cnt,
                            const float* __restrict__ bn1_g, const float* __restrict__ bn1_b,
                            const float* __restrict__ fc1_w, const float* __restrict__ fc1_b,
                            const float* __restrict__ bn2_g, const float* __restrict__ bn2_b,
                            const float* __restrict__ fc2_w, const float* __restrict__ fc2_b,
                            float* __restrict__ out) {
  int g = threadIdx.x;
  if (g >= GG) return;
  const float bn_s = rsqrtf(1.0f + 1e-5f);   // eval BN with running stats (0,1)
  float hc[256];
  float inv = 1.0f / fmaxf(cnt[g], 1.0f);
  for (int j = 0; j < 128; ++j) hc[j] = hm[g * 128 + j] * inv;
  for (int j = 0; j < 128; ++j) {
    float v = hx[g * 128 + j];
    hc[128 + j] = (fabsf(v) < 1e30f) ? v : 0.0f;
  }
  for (int j = 0; j < 256; ++j) hc[j] = hc[j] * bn_s * bn1_g[j] + bn1_b[j];
  float t[64];
  for (int o = 0; o < 64; ++o) {
    float a = fc1_b[o];
    for (int j = 0; j < 256; ++j) a += hc[j] * fc1_w[j * 64 + o];
    t[o] = eluf(a);
  }
  for (int o = 0; o < 64; ++o) t[o] = t[o] * bn_s * bn2_g[o] + bn2_b[o];
  for (int o = 0; o < 2; ++o) {
    float a = fc2_b[o];
    for (int j = 0; j < 64; ++j) a += t[j] * fc2_w[j * 2 + o];
    out[g * 2 + o] = a;
  }
}

// ------------------------------ host side ----------------------------------

static inline size_t al256(size_t x) { return (x + 255) & ~(size_t)255; }

extern "C" void kernel_launch(void* const* d_in, const int* in_sizes, int n_in,
                              void* d_out, int out_size, void* d_ws, size_t ws_size,
                              hipStream_t stream) {
  (void)in_sizes; (void)n_in; (void)out_size; (void)ws_size;

  const float* x      = (const float*)d_in[0];
  const int*   ei     = (const int*)  d_in[1];
  const int*   batch  = (const int*)  d_in[2];
  const float* W1     = (const float*)d_in[3];
  const float* a_src1 = (const float*)d_in[4];
  const float* a_dst1 = (const float*)d_in[5];
  const float* b1     = (const float*)d_in[6];
  const float* W2     = (const float*)d_in[7];
  const float* a_src2 = (const float*)d_in[8];
  const float* a_dst2 = (const float*)d_in[9];
  const float* b2     = (const float*)d_in[10];
  const float* W3     = (const float*)d_in[11];
  const float* b3     = (const float*)d_in[12];
  // d_in[13..16] = wq,bq,wk,bk: unused (seq_len==1 -> softmax over one key == 1)
  const float* wv     = (const float*)d_in[17];
  const float* bv     = (const float*)d_in[18];
  const float* wo     = (const float*)d_in[19];
  const float* bo     = (const float*)d_in[20];
  const float* ln1g   = (const float*)d_in[21];
  const float* ln1b   = (const float*)d_in[22];
  const float* ln2g   = (const float*)d_in[23];
  const float* ln2b   = (const float*)d_in[24];
  const float* wf1    = (const float*)d_in[25];
  const float* bf1    = (const float*)d_in[26];
  const float* wf2    = (const float*)d_in[27];
  const float* bf2    = (const float*)d_in[28];
  const float* bn1_g  = (const float*)d_in[29];
  const float* bn1_b  = (const float*)d_in[30];
  const float* fc1_w  = (const float*)d_in[31];
  const float* fc1_b  = (const float*)d_in[32];
  const float* bn2_g  = (const float*)d_in[33];
  const float* bn2_b  = (const float*)d_in[34];
  const float* fc2_w  = (const float*)d_in[35];
  const float* fc2_b  = (const float*)d_in[36];
  float* out = (float*)d_out;

  // ---- workspace arenas ----
  char* base = (char*)d_ws;
  size_t off = 0;
  auto carve = [&](size_t bytes) { char* p = base + off; off += al256(bytes); return p; };
  float*  bufA = (float*) carve((size_t)NN * 1024 * 4);  // GEMM outputs / temps
  float*  bufB = (float*) carve((size_t)NN * 1024 * 4);  // aggregated features / h
  __bf16* bufC = (__bf16*)carve((size_t)NN * 1024 * 2);  // bf16 activation staging
  __bf16* wbf  = (__bf16*)carve((size_t)1024 * 512 * 2); // bf16 weight staging
  float*  eW   = (float*) carve((size_t)ELP * 8 * 4);    // edge scores / exp
  float*  mW   = (float*) carve((size_t)NN * 8 * 4);     // segment max
  float*  sW   = (float*) carve((size_t)NN * 8 * 4);     // segment sum
  float*  aS   = (float*) carve((size_t)NN * 8 * 4);     // alpha_src
  float*  aD   = (float*) carve((size_t)NN * 8 * 4);     // alpha_dst
  float*  deg  = (float*) carve((size_t)NN * 4);
  float*  dinv = (float*) carve((size_t)NN * 4);
  float*  cnt  = (float*) carve((size_t)GG * 4);
  float*  hm   = (float*) carve((size_t)GG * 128 * 4);
  float*  hx   = (float*) carve((size_t)GG * 128 * 4);

  auto fill = [&](float* p, float v, long n) {
    fill_kernel<<<(unsigned)((n + 255) / 256), 256, 0, stream>>>(p, v, n);
  };
  auto cast = [&](const float* in, __bf16* o, long n) {
    cast_bf16_kernel<<<(unsigned)((n + 255) / 256), 256, 0, stream>>>(in, o, n);
  };
  auto gemm = [&](const __bf16* A, const __bf16* B, const float* bias, float* C,
                  int M, int K, int Nc, int act) {
    dim3 grid((M + 127) / 128, Nc / 64);
    gemm_bf16_wmma_kernel<<<grid, 256, 0, stream>>>(A, B, bias, C, M, K, Nc, act);
  };
  auto gat = [&](const float* hfeat, const float* as_w, const float* ad_w,
                 float* outAgg, int H) {
    gat_alpha_kernel<<<(NN * H + 255) / 256, 256, 0, stream>>>(hfeat, as_w, ad_w, aS, aD, H);
    fill(mW, -3.0e38f, (long)NN * H);
    fill(sW, 0.f,      (long)NN * H);
    edge_score_kernel<<<((long)ELP * H + 255) / 256, 256, 0, stream>>>(ei, H, aS, aD, eW, mW);
    edge_exp_kernel  <<<((long)ELP * H + 255) / 256, 256, 0, stream>>>(ei, H, eW, mW, sW);
    fill(outAgg, 0.f, (long)NN * H * HIDD);
    edge_aggr_kernel<<<ELP, 256, 0, stream>>>(ei, H, eW, sW, hfeat, outAgg);
  };

  // ================= GAT layer 1 (8 heads) =================
  cast(x, bufC, (long)NN * 128);
  cast(W1, wbf, 128L * 1024);
  gemm(bufC, wbf, nullptr, bufA, NN, 128, 1024, 0);     // h = x @ W1   [N,1024]
  gat(bufA, a_src1, a_dst1, bufB, 8);                   // aggregate -> bufB
  bias_elu_kernel<<<((long)NN * 1024 + 255) / 256, 256, 0, stream>>>(bufB, b1, (long)NN * 1024, 1024);

  // ================= GAT layer 2 (4 heads) =================
  cast(bufB, bufC, (long)NN * 1024);
  cast(W2, wbf, 1024L * 512);
  gemm(bufC, wbf, nullptr, bufA, NN, 1024, 512, 0);     // h = h @ W2   [N,512]
  gat(bufA, a_src2, a_dst2, bufB, 4);
  bias_elu_kernel<<<((long)NN * 512 + 255) / 256, 256, 0, stream>>>(bufB, b2, (long)NN * 512, 512);

  // ================= GCN =================
  fill(deg, 0.f, NN);
  deg_kernel<<<(ELP + 255) / 256, 256, 0, stream>>>(ei, deg);
  dinv_kernel<<<(NN + 255) / 256, 256, 0, stream>>>(deg, dinv);
  cast(bufB, bufC, (long)NN * 512);
  cast(W3, wbf, 512L * 128);
  gemm(bufC, wbf, nullptr, bufA, NN, 512, 128, 0);      // hw = h @ W3  [N,128]
  fill(bufB, 0.f, (long)NN * 128);
  gcn_aggr_kernel<<<ELP, 128, 0, stream>>>(ei, dinv, bufA, bufB);
  bias_elu_kernel<<<((long)NN * 128 + 255) / 256, 256, 0, stream>>>(bufB, b3, (long)NN * 128, 128);

  // ================= Transformer (6 layers, seq_len=1) =================
  float* h  = bufB;                        // [N,128]
  float* t1 = bufA;                        // [N,128]
  float* t2 = bufA + (size_t)NN * 128;     // [N,128]
  float* f1 = bufA;                        // [N,512] (reused after sa consumed)
  float* f2 = bufA + (size_t)NN * 512;     // [N,128]
  for (int l = 0; l < LL; ++l) {
    // sa = (h@wv+bv)@wo+bo
    cast(h, bufC, (long)NN * 128);
    cast(wv + (size_t)l * 128 * 128, wbf, 128L * 128);
    gemm(bufC, wbf, bv + l * 128, t1, NN, 128, 128, 0);
    cast(t1, bufC, (long)NN * 128);
    cast(wo + (size_t)l * 128 * 128, wbf, 128L * 128);
    gemm(bufC, wbf, bo + l * 128, t2, NN, 128, 128, 0);
    ln_add_kernel<<<(NN + 7) / 8, 256, 0, stream>>>(h, t2, ln1g + l * 128, ln1b + l * 128);
    // ff = gelu(h@wf1+bf1)@wf2+bf2
    cast(h, bufC, (long)NN * 128);
    cast(wf1 + (size_t)l * 128 * 512, wbf, 128L * 512);
    gemm(bufC, wbf, bf1 + l * 512, f1, NN, 128, 512, 1);   // fused exact GELU
    cast(f1, bufC, (long)NN * 512);
    cast(wf2 + (size_t)l * 512 * 128, wbf, 512L * 128);
    gemm(bufC, wbf, bf2 + l * 128, f2, NN, 512, 128, 0);
    ln_add_kernel<<<(NN + 7) / 8, 256, 0, stream>>>(h, f2, ln2g + l * 128, ln2b + l * 128);
  }

  // ================= pool + head =================
  fill(cnt, 0.f, GG);
  fill(hm, 0.f, (long)GG * 128);
  fill(hx, -3.0e38f, (long)GG * 128);
  pool_kernel<<<NN, 128, 0, stream>>>(h, batch, hm, hx, cnt);
  head_kernel<<<1, 64, 0, stream>>>(hm, hx, cnt, bn1_g, bn1_b, fc1_w, fc1_b,
                                    bn2_g, bn2_b, fc2_w, fc2_b, out);
}
